// TimeMix_16200616641142
// MI455X (gfx1250) — matrix operand order repaired and checked by reading, not compiled, not run on hardware
//
#include <hip/hip_runtime.h>
#include <hip/hip_bf16.h>
#include <cstdint>

// ---- problem constants (match reference) ----
#define BATCH 8
#define TLEN  2048
#define HDIM  2048
#define MROWS (BATCH * TLEN)   // 16384
#define LN_EPS 1e-5f

typedef __attribute__((ext_vector_type(16))) __bf16 v16bf;
typedef __attribute__((ext_vector_type(8)))  __bf16 bf16x8;
typedef __attribute__((ext_vector_type(8)))  float  v8f;
typedef __attribute__((ext_vector_type(4)))  unsigned int u32x4;
typedef __attribute__((ext_vector_type(8)))  int i32x8;
typedef __attribute__((ext_vector_type(4)))  int i32x4;

union Frag { v16bf v; bf16x8 h[2]; };

// ---------------------------------------------------------------------------
// TDM: issue a tensor_load_to_lds for a 128-row x 32-elem bf16 tile
// (row stride 2048 elems) into LDS with 16B padding every 64B so the LDS
// row stride is 80B (= LDT*2).  D# built per ISA §8.
// ---------------------------------------------------------------------------
__device__ __forceinline__ void tdm_load_tile(uint32_t lds_byte_addr,
                                              const __bf16* gsrc) {
  const unsigned long long ga = (unsigned long long)(uintptr_t)gsrc;
  u32x4 g0;
  g0[0] = 1u;                                   // count=1 (valid), user mode
  g0[1] = lds_byte_addr;                        // lds_addr [63:32]
  g0[2] = (unsigned)(ga & 0xffffffffu);         // global_addr lo
  g0[3] = (unsigned)((ga >> 32) & 0x1ffffffu)   // global_addr hi
          | (2u << 30);                         // type=2 ("image")
  i32x8 g1 = {};
  // data_size=1 (2B) | pad_enable | pad_interval=3 (16 DW) | pad_amount=3 (4 DW)
  g1[0] = (1 << 16) | (1 << 20) | (3 << 22) | (3 << 25);
  g1[1] = (int)((2048u & 0xffffu) << 16);       // tensor_dim0 = 2048
  g1[2] = (int)((128u  & 0xffffu) << 16);       // tensor_dim1 = 128
  g1[3] = (int)(32u << 16);                     // tile_dim0 = 32
  g1[4] = 128;                                  // tile_dim1 = 128
  g1[5] = 2048;                                 // tensor_dim0_stride = 2048
  g1[6] = 0;
  g1[7] = 0;
  i32x4 gz = {};
#if defined(__clang_major__) && (__clang_major__ >= 23)
  i32x8 z8 = {};
  __builtin_amdgcn_tensor_load_to_lds(g0, g1, gz, gz, z8, 0);
#else
  __builtin_amdgcn_tensor_load_to_lds(g0, g1, gz, gz, 0);
#endif
}

__device__ __forceinline__ void async_load_b128_to_lds(uint32_t lds_byte_addr,
                                                       const void* gaddr) {
  asm volatile("global_load_async_to_lds_b128 %0, %1, off"
               :: "v"(lds_byte_addr), "v"(gaddr) : "memory");
}

// ---------------------------------------------------------------------------
// Kernel 1: fused LayerNorm -> bf16   (one block of 256 threads per row)
// ---------------------------------------------------------------------------
__global__ __launch_bounds__(256)
void ln_kernel(const float* __restrict__ x, const float* __restrict__ scale,
               const float* __restrict__ bias, __bf16* __restrict__ xn) {
  const int row = blockIdx.x;                 // b*T + t
  const float* xr = x + (size_t)row * HDIM;
  __bf16* xo = xn + (size_t)row * HDIM;
  const int tid = threadIdx.x;

  float vals[8];
  float s = 0.f, ss = 0.f;
#pragma unroll
  for (int i = 0; i < 8; ++i) {
    float v = xr[tid + i * 256];
    vals[i] = v; s += v; ss += v * v;
  }
#pragma unroll
  for (int off = 16; off > 0; off >>= 1) {
    s  += __shfl_xor(s,  off, 32);
    ss += __shfl_xor(ss, off, 32);
  }
  __shared__ float red[16];
  const int wave = tid >> 5, lane = tid & 31;
  if (lane == 0) { red[wave] = s; red[wave + 8] = ss; }
  __syncthreads();
  float st = 0.f, sst = 0.f;
#pragma unroll
  for (int wv = 0; wv < 8; ++wv) { st += red[wv]; sst += red[wv + 8]; }
  const float mean = st * (1.0f / HDIM);
  const float var  = sst * (1.0f / HDIM) - mean * mean;
  const float inv  = rsqrtf(var + LN_EPS);
#pragma unroll
  for (int i = 0; i < 8; ++i) {
    const int c = tid + i * 256;
    xo[c] = (__bf16)((vals[i] - mean) * inv * scale[c] + bias[c]);
  }
}

// ---------------------------------------------------------------------------
// Kernel 2: pack weight W[k][n] (fp32 row-major) -> Wp[n][k] (bf16)
// ---------------------------------------------------------------------------
__global__ __launch_bounds__(256)
void packw_kernel(const float* __restrict__ W, __bf16* __restrict__ Wp) {
  __shared__ float tile[32][33];
  const int bx = blockIdx.x * 32, by = blockIdx.y * 32;
  const int tx = threadIdx.x, ty = threadIdx.y;  // 32 x 8
#pragma unroll
  for (int i = 0; i < 32; i += 8)
    tile[ty + i][tx] = W[(size_t)(by + ty + i) * HDIM + bx + tx];
  __syncthreads();
#pragma unroll
  for (int i = 0; i < 32; i += 8)
    Wp[(size_t)(bx + ty + i) * HDIM + by + tx] = (__bf16)tile[tx][ty + i];
}

// ---------------------------------------------------------------------------
// Kernel 3: double-buffered WMMA bf16 GEMM, 128x128x32 tiles, pipelined:
//   tile k+1 staged (TDM B / async or VALU-mix A) while tile k computes;
//   drained with partial waits s_wait_tensorcnt 1 / s_wait_asynccnt 2.
//   EPI: 0 = plain f32, 1 = sigmoid f32, 2 = residual-add f32
// ---------------------------------------------------------------------------
template<int EPI, bool MIX>
__global__ __launch_bounds__(256)
void gemm_kernel(const __bf16* __restrict__ A,
                 const float*  __restrict__ mix,
                 const __bf16* __restrict__ Bp,     // [N][K] bf16
                 float* __restrict__ Cf,
                 const float* __restrict__ resid) {
  constexpr int BM = 128, BN = 128, BK = 32, LDT = 40;  // 80B LDS row stride
  constexpr int NT = HDIM / BK;
  constexpr uint32_t BUFB = BM * LDT * 2;               // bytes per buffer
  __shared__ __attribute__((aligned(16))) __bf16 As[2][BM * LDT];
  __shared__ __attribute__((aligned(16))) __bf16 Bs[2][BN * LDT];

  const int tid  = threadIdx.x;
  const int wave = tid >> 5, lane = tid & 31;
  const int wm = wave >> 1, wn = wave & 1;     // 4 x 2 wave grid
  const int m16 = lane & 15, hi = lane >> 4;
  const int tileM = blockIdx.x * BM, tileN = blockIdx.y * BN;
  const uint32_t ldsA = (uint32_t)(uintptr_t)&As[0][0];
  const uint32_t ldsB = (uint32_t)(uintptr_t)&Bs[0][0];

  v8f acc[2][4] = {};

  // stage A tile for K offset k0 into buffer `buf`
  auto stageA = [&](int k0, int buf) {
#pragma unroll
    for (int it = 0; it < 2; ++it) {
      const int idx = tid + it * 256;          // 0..511
      const int row = idx >> 2;                // 0..127
      const int ch  = idx & 3;                 // 16B chunk within 64B row
      const int kk  = k0 + ch * 8;
      const int gm  = tileM + row;
      if (MIX) {
        const int t = gm & (TLEN - 1);
        bf16x8 c8 = *(const bf16x8*)(A + (size_t)gm * HDIM + kk);
        bf16x8 p8 = {};
        if (t > 0) p8 = *(const bf16x8*)(A + (size_t)(gm - 1) * HDIM + kk);
        const float4 m0 = *(const float4*)(mix + kk);
        const float4 m1 = *(const float4*)(mix + kk + 4);
        const float mv[8] = {m0.x, m0.y, m0.z, m0.w, m1.x, m1.y, m1.z, m1.w};
        bf16x8 o;
#pragma unroll
        for (int e = 0; e < 8; ++e) {
          const float c = (float)c8[e], p = (float)p8[e];
          o[e] = (__bf16)(c * mv[e] + p * (1.f - mv[e]));
        }
        *(bf16x8*)&As[buf][row * LDT + ch * 8] = o;
      } else {
        async_load_b128_to_lds(
            ldsA + (uint32_t)buf * BUFB + (uint32_t)(row * LDT + ch * 8) * 2,
            A + (size_t)gm * HDIM + kk);
      }
    }
  };

  // ---- prologue: stage tile 0 into buffer 0 ----
  if (wave == 0) tdm_load_tile(ldsB, Bp + (size_t)tileN * HDIM);
  stageA(0, 0);

  for (int kt = 0; kt < NT; ++kt) {
    const int cur = kt & 1, nxt = cur ^ 1;
    const bool has_next = (kt + 1 < NT);

    // ---- stage tile kt+1 into the other buffer (stays in flight) ----
    if (has_next) {
      const int k1 = (kt + 1) * BK;
      if (wave == 0)
        tdm_load_tile(ldsB + (uint32_t)nxt * BUFB,
                      Bp + (size_t)tileN * HDIM + k1);
      stageA(k1, nxt);
    }

    // ---- drain only tile kt (partial counter waits) ----
    if (!MIX) {
      if (has_next) asm volatile("s_wait_asynccnt 0x2" ::: "memory");
      else          asm volatile("s_wait_asynccnt 0x0" ::: "memory");
    }
    if (wave == 0) {
      if (has_next) __builtin_amdgcn_s_wait_tensorcnt(1);
      else          __builtin_amdgcn_s_wait_tensorcnt(0);
    }
    __syncthreads();

    // ---- fragments per ISA bf16 layouts ----
    Frag af[2], bfr[4];
#pragma unroll
    for (int mt = 0; mt < 2; ++mt) {
      const int r = wm * 32 + mt * 16 + m16;        // lane -> row M
      af[mt].h[0] = *(const bf16x8*)&As[cur][r * LDT + 8 * hi];
      af[mt].h[1] = *(const bf16x8*)&As[cur][r * LDT + 16 + 8 * hi];
    }
#pragma unroll
    for (int nt = 0; nt < 4; ++nt) {
      const int r = wn * 64 + nt * 16 + m16;        // lane -> col N
      bfr[nt].h[0] = *(const bf16x8*)&Bs[cur][r * LDT + 8 * hi];
      bfr[nt].h[1] = *(const bf16x8*)&Bs[cur][r * LDT + 16 + 8 * hi];
    }
#pragma unroll
    for (int mt = 0; mt < 2; ++mt)
#pragma unroll
      for (int nt = 0; nt < 4; ++nt)
        acc[mt][nt] = __builtin_amdgcn_wmma_f32_16x16x32_bf16(
            false, af[mt].v, false, bfr[nt].v, (short)0, acc[mt][nt],
            false, false);
    __syncthreads();   // protect buffer `cur` from next iteration's staging
  }

  // ---- epilogue: C layout VGPR r -> M = r + 8*hi, lane -> N ----
#pragma unroll
  for (int mt = 0; mt < 2; ++mt) {
#pragma unroll
    for (int nt = 0; nt < 4; ++nt) {
      const int col = tileN + wn * 64 + nt * 16 + m16;
#pragma unroll
      for (int r = 0; r < 8; ++r) {
        const int rowg = tileM + wm * 32 + mt * 16 + r + 8 * hi;
        const size_t off = (size_t)rowg * HDIM + col;
        float v = acc[mt][nt][r];
        if (EPI == 1) v = 1.f / (1.f + __expf(-v));
        if (EPI == 2) v += resid[off];
        Cf[off] = v;
      }
    }
  }
}

// ---------------------------------------------------------------------------
// Kernel 4: numerically-stabilized WKV scan; one thread per (b,h) channel.
// ---------------------------------------------------------------------------
__global__ __launch_bounds__(256)
void wkv_kernel(const float* __restrict__ k, const float* __restrict__ v,
                const float* __restrict__ r, const float* __restrict__ tf,
                const float* __restrict__ td, __bf16* __restrict__ rwkv) {
  const int gid = blockIdx.x * blockDim.x + threadIdx.x;  // 0 .. B*H-1
  const int b = gid >> 11;             // / HDIM
  const int h = gid & (HDIM - 1);
  const float u = tf[h];
  const float w = -__expf(td[h]);
  float a = 0.f, bb = 0.f, p = 1.0f - 1e30f;
  const size_t base = (size_t)b * TLEN * HDIM + h;
  for (int t = 0; t < TLEN; ++t) {
    const size_t idx = base + (size_t)t * HDIM;
    if (t + 8 < TLEN) {
      __builtin_prefetch(k + idx + 8 * HDIM, 0, 1);
      __builtin_prefetch(v + idx + 8 * HDIM, 0, 1);
      __builtin_prefetch(r + idx + 8 * HDIM, 0, 1);
    }
    const float kt = k[idx], vt = v[idx];
    const float uk = kt + u;
    const float q  = fmaxf(uk, p);
    const float e1 = __expf(p - q), e2 = __expf(uk - q);
    const float num = e1 * a + e2 * vt;
    const float den = e1 * bb + e2;
    const float wkv = (den == 0.f) ? 0.f : num / den;
    rwkv[idx] = (__bf16)(r[idx] * wkv);
    const float cw = w + p;
    const float q2 = fmaxf(cw, kt);
    const float ce1 = __expf(cw - q2), ce2 = __expf(kt - q2);
    a = ce1 * a + ce2 * vt;
    bb = ce1 * bb + ce2;
    p = q2;
  }
}

// ---------------------------------------------------------------------------
extern "C" void kernel_launch(void* const* d_in, const int* in_sizes, int n_in,
                              void* d_out, int out_size, void* d_ws, size_t ws_size,
                              hipStream_t stream) {
  const float* x          = (const float*)d_in[0];
  const float* ln_scale   = (const float*)d_in[1];
  const float* ln_bias    = (const float*)d_in[2];
  const float* mix_k      = (const float*)d_in[3];
  const float* mix_v      = (const float*)d_in[4];
  const float* mix_r      = (const float*)d_in[5];
  const float* time_first = (const float*)d_in[6];
  const float* time_decay = (const float*)d_in[7];
  const float* Wk         = (const float*)d_in[8];
  const float* Wv         = (const float*)d_in[9];
  const float* Wr         = (const float*)d_in[10];
  const float* Wo         = (const float*)d_in[11];
  float* out = (float*)d_out;

  // ---- workspace carve-up ----
  char* ws = (char*)d_ws;
  __bf16* xn  = (__bf16*)ws; ws += (size_t)MROWS * HDIM * 2;   //  64 MB
  __bf16* Wkp = (__bf16*)ws; ws += (size_t)HDIM * HDIM * 2;    //   8 MB
  __bf16* Wvp = (__bf16*)ws; ws += (size_t)HDIM * HDIM * 2;
  __bf16* Wrp = (__bf16*)ws; ws += (size_t)HDIM * HDIM * 2;
  __bf16* Wop = (__bf16*)ws; ws += (size_t)HDIM * HDIM * 2;
  float* kbuf = (float*)ws;  ws += (size_t)MROWS * HDIM * 4;   // 128 MB
  float* vbuf = (float*)ws;  ws += (size_t)MROWS * HDIM * 4;
  float* rbuf = (float*)ws;  ws += (size_t)MROWS * HDIM * 4;
  __bf16* rwkv = (__bf16*)ws;                                  //  64 MB

  // 1) LayerNorm -> bf16
  ln_kernel<<<MROWS, 256, 0, stream>>>(x, ln_scale, ln_bias, xn);

  // 2) pack weights (transpose + bf16)
  dim3 pg(HDIM / 32, HDIM / 32), pb(32, 8);
  packw_kernel<<<pg, pb, 0, stream>>>(Wk, Wkp);
  packw_kernel<<<pg, pb, 0, stream>>>(Wv, Wvp);
  packw_kernel<<<pg, pb, 0, stream>>>(Wr, Wrp);
  packw_kernel<<<pg, pb, 0, stream>>>(Wo, Wop);

  // 3) three projection GEMMs with fused time-shift mixing
  dim3 gg(MROWS / 128, HDIM / 128), gb(256);
  gemm_kernel<0, true><<<gg, gb, 0, stream>>>(xn, mix_k, Wkp, kbuf, nullptr);
  gemm_kernel<0, true><<<gg, gb, 0, stream>>>(xn, mix_v, Wvp, vbuf, nullptr);
  gemm_kernel<1, true><<<gg, gb, 0, stream>>>(xn, mix_r, Wrp, rbuf, nullptr);

  // 4) WKV scan (writes r*wkv as bf16)
  wkv_kernel<<<(BATCH * HDIM) / 256, 256, 0, stream>>>(
      kbuf, vbuf, rbuf, time_first, time_decay, rwkv);

  // 5) output GEMM with fused residual add
  gemm_kernel<2, false><<<gg, gb, 0, stream>>>(rwkv, nullptr, Wop, out, x);
}
